// MLP_38723425141338
// MI455X (gfx1250) — matrix operand order repaired and checked by reading, not compiled, not run on hardware
//
#include <hip/hip_runtime.h>

// Fused NeRF-style MLP for MI455X (gfx1250, wave32, v_wmma_f32_16x16x32_f16).
// D = A x B with A = weights (out x in), B = batch columns (in x 16 rows).
// Batch stays "in lanes" through all layers; inter-layer transpose is a single
// lane<->lane+16 swap via ds_swizzle SWAPX16 on packed-f16 pairs.
// Next tile's input is prefetched into L2 (global_prefetch_b8, zero VGPR cost).

typedef __attribute__((ext_vector_type(16))) _Float16 v16h;
typedef __attribute__((ext_vector_type(8)))  _Float16 v8h;
typedef __attribute__((ext_vector_type(2)))  _Float16 v2h;
typedef __attribute__((ext_vector_type(8)))  float    v8f;
typedef __attribute__((ext_vector_type(4)))  float    v4f;

union V16HU { v16h v; v8h h[2]; int i[8]; };

#define WMMA_F16(A, B, C) \
  __builtin_amdgcn_wmma_f32_16x16x32_f16(false, (A), false, (B), (short)0, (C), false, false)

// group-of-32 swizzle, SWAPX16: xor=0x10, or=0, and=0x1f -> 0x401F
#define SWZ_SWAP16(x) __builtin_amdgcn_ds_swizzle((x), 0x401F)

#define NW 8  // waves per block (256 threads)

// pack two f32 into one VGPR holding two f16 (v_cvt_pk_f16_f32)
static __device__ __forceinline__ int pk_f16(float a, float b) {
  v2h p; p.x = (_Float16)a; p.y = (_Float16)b;
  return __builtin_bit_cast(int, p);
}
// ReLU in packed f16: cvt output is canonical -> one v_pk_max_num_f16
static __device__ __forceinline__ int pk_relu_f16(float a, float b) {
  v2h p; p.x = (_Float16)a; p.y = (_Float16)b;
  v2h z = {};
  p = __builtin_elementwise_max(p, z);
  return __builtin_bit_cast(int, p);
}

// A-fragment (16x32 f16, MxK) for weights stored row-major (out,in) in LDS.
// lanes 0-15 : row M=lane,    halfs = K{k0+0..15}
// lanes 16-31: row M=lane-16, halfs = K{k0+16..31}
static __device__ __forceinline__ v16h lds_load_A(const _Float16* base, int in_f, int k0, int lane) {
  const int r = lane & 15;
  const _Float16* p = base + r * in_f + k0 + ((lane & 16) ? 16 : 0);
  V16HU u;
  u.h[0] = *(const v8h*)(p);
  u.h[1] = *(const v8h*)(p + 8);
  return u.v;
}

// Build a 32x16 B fragment (K-chunk of 32 features) from two packed D tiles.
// ta = packed f16 pairs of features [base..base+15], tb = [base+16..base+31]
// (each lane holds 4 ints = 8 features; lanes<16 hold M 0..7, lanes>=16 M 8..15).
// B layout: lanes<16 hold K=0..15 (VGPR j = K 2j,2j+1); lanes>=16 hold K=16..31.
static __device__ __forceinline__ v16h build_B(const int* ta, const int* tb, int upper) {
  V16HU u;
#pragma unroll
  for (int j = 0; j < 4; ++j) {
    const int sa = SWZ_SWAP16(ta[j]);
    const int sb = SWZ_SWAP16(tb[j]);
    u.i[j]     = upper ? sb    : ta[j];
    u.i[4 + j] = upper ? tb[j] : sa;
  }
  return u.v;
}

__global__ __launch_bounds__(256) void nerf_fused_mlp(
    const float* __restrict__ x,
    const float* __restrict__ Ws0, const float* __restrict__ Ws1,
    const float* __restrict__ Wc0, const float* __restrict__ Wc1,
    const float* __restrict__ Wc2,
    float* __restrict__ out, int nrows)
{
  // f16 row-major weight copies; Wc2 padded 3->16 rows with zeros.
  __shared__ __align__(32) _Float16 s_ws0[64 * 32];
  __shared__ __align__(32) _Float16 s_ws1[16 * 64];
  __shared__ __align__(32) _Float16 s_wc0[64 * 32];
  __shared__ __align__(32) _Float16 s_wc1[64 * 64];
  __shared__ __align__(32) _Float16 s_wc2[16 * 64];

  const int tid   = threadIdx.x;
  const int lane  = tid & 31;
  const int wv    = tid >> 5;
  const int upper = lane & 16;

  for (int i = tid; i < 64 * 32; i += 256) s_ws0[i] = (_Float16)Ws0[i];
  for (int i = tid; i < 16 * 64; i += 256) s_ws1[i] = (_Float16)Ws1[i];
  for (int i = tid; i < 64 * 32; i += 256) s_wc0[i] = (_Float16)Wc0[i];
  for (int i = tid; i < 64 * 64; i += 256) s_wc1[i] = (_Float16)Wc1[i];
  for (int i = tid; i < 16 * 64; i += 256) s_wc2[i] = (i < 3 * 64) ? (_Float16)Wc2[i] : (_Float16)0.0f;
  __syncthreads();

  // ---- hoist all weight A-fragments into registers (loop-invariant) ----
  v16h ws0f[4], ws1f[2], wc0f[4], wc1f[8], wc2f[2];
#pragma unroll
  for (int f = 0; f < 4; ++f) ws0f[f] = lds_load_A(s_ws0 + f * 16 * 32, 32, 0, lane);
  ws1f[0] = lds_load_A(s_ws1, 64, 0, lane);
  ws1f[1] = lds_load_A(s_ws1, 64, 32, lane);
#pragma unroll
  for (int f = 0; f < 4; ++f) wc0f[f] = lds_load_A(s_wc0 + f * 16 * 32, 32, 0, lane);
#pragma unroll
  for (int f = 0; f < 4; ++f) {
    wc1f[2 * f + 0] = lds_load_A(s_wc1 + f * 16 * 64, 64, 0, lane);
    wc1f[2 * f + 1] = lds_load_A(s_wc1 + f * 16 * 64, 64, 32, lane);
  }
  wc2f[0] = lds_load_A(s_wc2, 64, 0, lane);
  wc2f[1] = lds_load_A(s_wc2, 64, 32, lane);

  const int ntiles  = nrows >> 4;
  const int strideT = (int)gridDim.x * NW;
  // per-lane, loop-invariant offsets (32-bit: max x offset ~403MB < 2^31 bytes)
  const int rl      = lane & 15;
  const int xoffPos = rl * 48 + 16 + (upper ? 16 : 0);  // floats
  const int xoffDir = rl * 48;                          // floats

  for (int t = (int)blockIdx.x * NW + wv; t < ntiles; t += strideT) {
    const int row0  = t << 4;
    const int xbase = row0 * 48;  // floats

    // ---- prefetch next tile's input into L2 (zero register cost) ----
    if (t + strideT < ntiles) {  // wave-uniform; EXEC stays full
      const int nbase = (t + strideT) * (16 * 48);
      __builtin_prefetch(x + nbase + xoffPos, 0, 1);
      __builtin_prefetch(x + nbase + xoffDir, 0, 1);
    }

    // ---- B fragment for pos (x cols 16..47): lane n reads 16 consecutive
    // floats (cols 16..31 lower half / 32..47 upper half), 64B aligned ----
    const v4f* pp = (const v4f*)(x + xbase + xoffPos);
    V16HU bx;
#pragma unroll
    for (int q = 0; q < 4; ++q) {
      const v4f f = pp[q];
      bx.i[2 * q + 0] = pk_f16(f.x, f.y);
      bx.i[2 * q + 1] = pk_f16(f.z, f.w);
    }
    // dir (x cols 0..15) packed; lanes n and n+16 read the same 64B (cache hit)
    const v4f* pd = (const v4f*)(x + xbase + xoffDir);
    int dirp[8];
#pragma unroll
    for (int q = 0; q < 4; ++q) {
      const v4f f = pd[q];
      dirp[2 * q + 0] = pk_f16(f.x, f.y);
      dirp[2 * q + 1] = pk_f16(f.z, f.w);
    }

    // ---- sigma L0: h = relu(Ws0 . pos^T), 64 features = 4 D tiles ----
    int hp[4][4];
#pragma unroll
    for (int f = 0; f < 4; ++f) {
      v8f z = {};
      const v8f acc = WMMA_F16(ws0f[f], bx.v, z);
#pragma unroll
      for (int j = 0; j < 4; ++j) hp[f][j] = pk_relu_f16(acc[2 * j], acc[2 * j + 1]);
    }

    // ---- sigma L1: o = Ws1 . h^T (K=64 -> two B chunks) ----
    v8f oacc = {};
    oacc = WMMA_F16(ws1f[0], build_B(hp[0], hp[1], upper), oacc);
    oacc = WMMA_F16(ws1f[1], build_B(hp[2], hp[3], upper), oacc);
    const float sig0 = oacc[0];  // feature 0 lives in lanes<16

    int op[4];
#pragma unroll
    for (int j = 0; j < 4; ++j) op[j] = pk_f16(oacc[2 * j], oacc[2 * j + 1]);

    // ---- color input B fragment: [dir(16) | s1-out(16)] ----
    V16HU bci;
#pragma unroll
    for (int j = 0; j < 4; ++j) {
      const int s = SWZ_SWAP16(op[j]);
      bci.i[j]     = upper ? s     : dirp[j];
      bci.i[4 + j] = upper ? op[j] : dirp[4 + j];
    }

    // ---- color L0: c = relu(Wc0 . ci^T) ----
    int cp[4][4];
#pragma unroll
    for (int f = 0; f < 4; ++f) {
      v8f z = {};
      const v8f acc = WMMA_F16(wc0f[f], bci.v, z);
#pragma unroll
      for (int j = 0; j < 4; ++j) cp[f][j] = pk_relu_f16(acc[2 * j], acc[2 * j + 1]);
    }

    // ---- color L1: c = relu(Wc1 . c^T) (K=64) ----
    const v16h b0 = build_B(cp[0], cp[1], upper);
    const v16h b1 = build_B(cp[2], cp[3], upper);
    int dp[4][4];
#pragma unroll
    for (int f = 0; f < 4; ++f) {
      v8f z = {};
      v8f acc = WMMA_F16(wc1f[2 * f + 0], b0, z);
      acc     = WMMA_F16(wc1f[2 * f + 1], b1, acc);
#pragma unroll
      for (int j = 0; j < 4; ++j) dp[f][j] = pk_relu_f16(acc[2 * j], acc[2 * j + 1]);
    }

    // ---- color L2: rgb = Wc2 . c^T (K=64, rows 0..2 valid) ----
    const v16h e0 = build_B(dp[0], dp[1], upper);
    const v16h e1 = build_B(dp[2], dp[3], upper);
    v8f z = {};
    v8f col = WMMA_F16(wc2f[0], e0, z);
    col     = WMMA_F16(wc2f[1], e1, col);

    // ---- output: lane n (<16) owns batch row row0+n completely ----
    if (!upper) {
      v4f o;
      o.x = 1.0f / (1.0f + __expf(-col[0]));
      o.y = 1.0f / (1.0f + __expf(-col[1]));
      o.z = 1.0f / (1.0f + __expf(-col[2]));
      o.w = __expf(sig0);
      *(v4f*)(out + (row0 + lane) * 4) = o;
    }
  }
}

extern "C" void kernel_launch(void* const* d_in, const int* in_sizes, int n_in,
                              void* d_out, int out_size, void* d_ws, size_t ws_size,
                              hipStream_t stream) {
  const float* x   = (const float*)d_in[0];
  const float* Ws0 = (const float*)d_in[1];
  const float* Ws1 = (const float*)d_in[2];
  const float* Wc0 = (const float*)d_in[3];
  const float* Wc1 = (const float*)d_in[4];
  const float* Wc2 = (const float*)d_in[5];
  float* out = (float*)d_out;

  const int nrows  = in_sizes[0] / 48;   // 2,097,152
  const int ntiles = nrows / 16;         // 131,072 tiles
  int blocks = 2048;                     // 16,384 waves -> 8 tiles/wave
  const int wavesNeeded = (ntiles + NW - 1) / NW;
  if (blocks > wavesNeeded) blocks = wavesNeeded > 0 ? wavesNeeded : 1;

  nerf_fused_mlp<<<blocks, 256, 0, stream>>>(x, Ws0, Ws1, Wc0, Wc1, Wc2, out, nrows);
}
